// GlobalModel_ONE_20839181320244
// MI455X (gfx1250) — matrix operand order repaired and checked by reading, not compiled, not run on hardware
//
#include <hip/hip_runtime.h>
#include <hip/hip_bf16.h>

// ---------------------------------------------------------------------------
// CDNA5 (gfx1250) wave32 WMMA types
// ---------------------------------------------------------------------------
typedef __attribute__((ext_vector_type(16))) _Float16 v16h;
typedef __attribute__((ext_vector_type(8)))  _Float16 v8h;
typedef __attribute__((ext_vector_type(8)))  float    v8f;

__device__ __forceinline__ v8f wmma32(v16h a, v16h b, v8f c) {
  // D = A(16x32 f16) * B(32x16 f16) + C(16x16 f32)
  return __builtin_amdgcn_wmma_f32_16x16x32_f16(
      /*neg_a=*/false, a, /*neg_b=*/false, b,
      /*c_mod=*/(short)0, c, /*reuse_a=*/false, /*reuse_b=*/false);
}

// A-fragment load from an f32 row in global memory.
// k order used here (contiguous, not AMD's table order): lane L (row m=L&15, kh=L>>4) holds
// K = [c0, c0+16) where c0 = ktile*32 + kh*16.
__device__ __forceinline__ v16h load_a_f32(const float* __restrict__ p, int c0) {
  const float4* q0 = reinterpret_cast<const float4*>(p + c0);
  const float4* q1 = reinterpret_cast<const float4*>(p + c0 + 8);
  float4 u0 = q0[0], u1 = q0[1], u2 = q1[0], u3 = q1[1];
  v16h a;
  a[0]  = (_Float16)u0.x; a[1]  = (_Float16)u0.y; a[2]  = (_Float16)u0.z; a[3]  = (_Float16)u0.w;
  a[4]  = (_Float16)u1.x; a[5]  = (_Float16)u1.y; a[6]  = (_Float16)u1.z; a[7]  = (_Float16)u1.w;
  a[8]  = (_Float16)u2.x; a[9]  = (_Float16)u2.y; a[10] = (_Float16)u2.z; a[11] = (_Float16)u2.w;
  a[12] = (_Float16)u3.x; a[13] = (_Float16)u3.y; a[14] = (_Float16)u3.z; a[15] = (_Float16)u3.w;
  return a;
}

// A-fragment load from an f16 row staged in LDS (two 16B ds reads).
__device__ __forceinline__ v16h load_a_f16(const _Float16* p, int c0) {
  v8h lo = *reinterpret_cast<const v8h*>(p + c0);
  v8h hi = *reinterpret_cast<const v8h*>(p + c0 + 8);
  v16h a;
#pragma unroll
  for (int i = 0; i < 8; ++i) { a[i] = lo[i]; a[8 + i] = hi[i]; }
  return a;
}

__device__ __forceinline__ float leaky(float v) { return v >= 0.f ? v : 0.01f * v; }

// ---------------------------------------------------------------------------
// k_zero: clear accumulators (stat buffers + gx + ga) each launch
// ---------------------------------------------------------------------------
__global__ void k_zero(float* __restrict__ p, int n) {
  int i = blockIdx.x * 256 + threadIdx.x;
  if (i < n) p[i] = 0.f;
}

// ---------------------------------------------------------------------------
// k_prepW1: W1a (128x128 f32, row-major [k][j]) -> WT1 f16 transposed [j][k]
// so B-fragments are 16 contiguous halves per lane.
// ---------------------------------------------------------------------------
__global__ void k_prepW1(const float* __restrict__ W1a, _Float16* __restrict__ WT1) {
  int id = blockIdx.x * 256 + threadIdx.x;
  if (id < 128 * 128) {
    int k = id >> 7, j = id & 127;
    WT1[j * 128 + k] = (_Float16)W1a[id];
  }
}

// ---------------------------------------------------------------------------
// k_edge1: layer-1 of edge MLP via WMMA; accumulate per-feature sum / sumsq
// of leaky(concat(x[row],edge_attr) @ W1a + b1a) for the batch-norm stats.
// 8 waves/block, one 16-edge tile per wave per iteration.
// nt loop capped at unroll 2: keeps VGPR usage < 256 (no s_set_vgpr_msb
// churn) and lets several waves/SIMD stay resident to hide gather latency.
// ---------------------------------------------------------------------------
__global__ void __launch_bounds__(256) k_edge1(
    const float* __restrict__ x, const int* __restrict__ eidx,
    const float* __restrict__ eattr, const float* __restrict__ b1a,
    const _Float16* __restrict__ WT1, float* __restrict__ sumBuf,
    float* __restrict__ sqBuf, int E) {
  __shared__ _Float16 sW[128 * 128];  // W1a^T f16 (32 KB)
  __shared__ float sB1[128];          // bias
  __shared__ float acc[256];          // per-feature sum(128) + sumsq(128)
  const int tid = threadIdx.x;
  for (int i = tid; i < 2048; i += 256)
    reinterpret_cast<uint4*>(sW)[i] = reinterpret_cast<const uint4*>(WT1)[i];
  if (tid < 128) sB1[tid] = b1a[tid];
  acc[tid & 255] = 0.f;
  __syncthreads();

  const int wave = tid >> 5, lane = tid & 31;
  const int mrow = lane & 15, kh = lane >> 4;
  const int ntiles = (E + 15) >> 4;

  for (int tile = blockIdx.x * 8 + wave; tile < ntiles; tile += gridDim.x * 8) {
    int e = tile * 16 + mrow;
    if (e >= E) e = E - 1;  // E is a multiple of 16 in practice
    const float* xr = x + (size_t)eidx[e] * 64;      // gathered source node
    const float* ar = eattr + (size_t)e * 64;        // edge features
    v16h A0 = load_a_f32(xr, kh * 16);               // K   0..31  (x part)
    v16h A1 = load_a_f32(xr, 32 + kh * 16);          // K  32..63
    v16h A2 = load_a_f32(ar, kh * 16);               // K  64..95  (edge part)
    v16h A3 = load_a_f32(ar, 32 + kh * 16);          // K  96..127

#pragma unroll 2
    for (int nt = 0; nt < 8; ++nt) {
      v8f c = {};
      const _Float16* wb = sW + (nt * 16 + mrow) * 128 + kh * 16;
      c = wmma32(A0, *reinterpret_cast<const v16h*>(wb),      c);
      c = wmma32(A1, *reinterpret_cast<const v16h*>(wb + 32), c);
      c = wmma32(A2, *reinterpret_cast<const v16h*>(wb + 64), c);
      c = wmma32(A3, *reinterpret_cast<const v16h*>(wb + 96), c);
      float bias = sB1[nt * 16 + mrow];
      float s = 0.f, s2 = 0.f;
#pragma unroll
      for (int r = 0; r < 8; ++r) {
        float v = leaky(c[r] + bias);
        s += v; s2 += v * v;
      }
      atomicAdd(&acc[nt * 16 + mrow], s);
      atomicAdd(&acc[128 + nt * 16 + mrow], s2);
    }
  }
  __syncthreads();
  if (tid < 128) {
    atomicAdd(&sumBuf[tid], acc[tid]);
    atomicAdd(&sqBuf[tid],  acc[128 + tid]);
  }
}

// ---------------------------------------------------------------------------
// k_fold1: BN(mean/var over E) folded into W2a.
//   s[k] = g1a[k]*rsqrt(var+eps);  o[k] = be1a[k] - mean[k]*s[k]
//   W2T[j][k] = f16(s[k]*W2a[k][j]);  b2P[j] = b2a[j] + sum_k o[k]*W2a[k][j]
// ---------------------------------------------------------------------------
__global__ void k_fold1(const float* __restrict__ sumBuf, const float* __restrict__ sqBuf,
                        const float* __restrict__ g1a, const float* __restrict__ be1a,
                        const float* __restrict__ W2a, const float* __restrict__ b2a,
                        float invE, _Float16* __restrict__ W2T, float* __restrict__ b2P) {
  __shared__ float s[128], o[128];
  int j = threadIdx.x;
  float m  = sumBuf[j] * invE;
  float v  = sqBuf[j] * invE - m * m;
  float sc = g1a[j] * rsqrtf(v + 1e-5f);
  s[j] = sc;
  o[j] = be1a[j] - m * sc;
  __syncthreads();
  float acc = b2a[j];
  for (int k = 0; k < 128; ++k) {
    float w = W2a[k * 128 + j];         // coalesced across threads
    acc += o[k] * w;
    W2T[j * 128 + k] = (_Float16)(s[k] * w);
  }
  b2P[j] = acc;
}

// ---------------------------------------------------------------------------
// k_gx: per-graph sums of node features (segment_sum x by batch)
// ---------------------------------------------------------------------------
__global__ void k_gx(const float* __restrict__ x, const int* __restrict__ batch,
                     float* __restrict__ gx, int N) {
  int id = blockIdx.x * 256 + threadIdx.x;
  if (id < N * 64) {
    int n = id >> 6, f = id & 63;
    atomicAdd(&gx[(size_t)batch[n] * 64 + f], x[id]);
  }
}

// ---------------------------------------------------------------------------
// k_edge2: recompute layer-1 fragments, LDS transpose C-frag -> A-frag,
// layer-2 WMMA with BN-folded weights, scatter-add into ga[batch[col[e]]].
// LDS: 32 KB W1 + 32 KB W2 + 32 KB tiles -> 3 blocks/WGP, 6 waves/SIMD.
// ---------------------------------------------------------------------------
__global__ void __launch_bounds__(256) k_edge2(
    const float* __restrict__ x, const int* __restrict__ eidx,
    const float* __restrict__ eattr, const int* __restrict__ batch,
    const float* __restrict__ b1a, const _Float16* __restrict__ WT1,
    const _Float16* __restrict__ W2T, const float* __restrict__ b2P,
    float* __restrict__ ga, int E) {
  __shared__ _Float16 sW1[128 * 128];   // 32 KB
  __shared__ _Float16 sW2[128 * 128];   // 32 KB
  __shared__ _Float16 sT[8][16 * 128];  // 32 KB: per-wave 16x128 f16 tile
  __shared__ float sB1[128], sB2[128];
  __shared__ int sBid[8][16];
  const int tid = threadIdx.x;
  for (int i = tid; i < 2048; i += 256) {
    reinterpret_cast<uint4*>(sW1)[i] = reinterpret_cast<const uint4*>(WT1)[i];
    reinterpret_cast<uint4*>(sW2)[i] = reinterpret_cast<const uint4*>(W2T)[i];
  }
  if (tid < 128) { sB1[tid] = b1a[tid]; sB2[tid] = b2P[tid]; }
  __syncthreads();

  const int wave = tid >> 5, lane = tid & 31;
  const int mrow = lane & 15, kh = lane >> 4;
  _Float16* tT = sT[wave];
  const int ntiles = (E + 15) >> 4;

  for (int tile = blockIdx.x * 8 + wave; tile < ntiles; tile += gridDim.x * 8) {
    if (lane < 16) {
      int ee = tile * 16 + lane;
      sBid[wave][lane] = (ee < E) ? batch[eidx[E + ee]] : -1;  // col -> graph id
    }
    __builtin_amdgcn_wave_barrier();

    int e = tile * 16 + mrow;
    if (e >= E) e = E - 1;
    const float* xr = x + (size_t)eidx[e] * 64;
    const float* ar = eattr + (size_t)e * 64;
    v16h A0 = load_a_f32(xr, kh * 16);
    v16h A1 = load_a_f32(xr, 32 + kh * 16);
    v16h A2 = load_a_f32(ar, kh * 16);
    v16h A3 = load_a_f32(ar, 32 + kh * 16);

    // ---- layer 1 (recompute), stage t = leaky(.) as f16 tile in LDS ----
#pragma unroll 2
    for (int nt = 0; nt < 8; ++nt) {
      v8f c = {};
      const _Float16* wb = sW1 + (nt * 16 + mrow) * 128 + kh * 16;
      c = wmma32(A0, *reinterpret_cast<const v16h*>(wb),      c);
      c = wmma32(A1, *reinterpret_cast<const v16h*>(wb + 32), c);
      c = wmma32(A2, *reinterpret_cast<const v16h*>(wb + 64), c);
      c = wmma32(A3, *reinterpret_cast<const v16h*>(wb + 96), c);
      float bias = sB1[nt * 16 + mrow];
#pragma unroll
      for (int r = 0; r < 8; ++r) {
        // C layout: VGPR r holds rows m=r (lanes 0-15), m=8+r (lanes 16-31)
        tT[(r + 8 * kh) * 128 + nt * 16 + mrow] = (_Float16)leaky(c[r] + bias);
      }
    }
    __builtin_amdgcn_wave_barrier();  // wave-local LDS ordering (DS in-order)

    // ---- layer 2: t @ (BN-folded W2a) + b2P, scatter into ga ----
    const _Float16* trow = tT + mrow * 128;
    v16h T0 = load_a_f16(trow, kh * 16);
    v16h T1 = load_a_f16(trow, 32 + kh * 16);
    v16h T2 = load_a_f16(trow, 64 + kh * 16);
    v16h T3 = load_a_f16(trow, 96 + kh * 16);

#pragma unroll 2
    for (int nt = 0; nt < 8; ++nt) {
      v8f c = {};
      const _Float16* wb = sW2 + (nt * 16 + mrow) * 128 + kh * 16;
      c = wmma32(T0, *reinterpret_cast<const v16h*>(wb),      c);
      c = wmma32(T1, *reinterpret_cast<const v16h*>(wb + 32), c);
      c = wmma32(T2, *reinterpret_cast<const v16h*>(wb + 64), c);
      c = wmma32(T3, *reinterpret_cast<const v16h*>(wb + 96), c);
      float bias = sB2[nt * 16 + mrow];
#pragma unroll
      for (int r = 0; r < 8; ++r) {
        int b = sBid[wave][r + 8 * kh];
        if (b >= 0)
          atomicAdd(&ga[(size_t)b * 128 + nt * 16 + mrow], c[r] + bias);
      }
    }
    __builtin_amdgcn_wave_barrier();  // before next iter reuses tT/sBid
  }
}

// ---------------------------------------------------------------------------
// Tiny per-graph MLP (B=500 rows): plain f32 — 0.05% of total FLOPs
// ---------------------------------------------------------------------------
__global__ void __launch_bounds__(128) k_g1(
    const float* __restrict__ gx, const float* __restrict__ ga,
    const float* __restrict__ W1b, const float* __restrict__ b1b,
    float* __restrict__ a1, float* __restrict__ sum2, float* __restrict__ sq2) {
  int i = blockIdx.x, j = threadIdx.x;
  const float* gxr = gx + (size_t)i * 64;
  const float* gar = ga + (size_t)i * 128;
  float acc = b1b[j];
  for (int k = 0; k < 64; ++k)  acc += gxr[k] * W1b[k * 128 + j];
  for (int k = 0; k < 128; ++k) acc += gar[k] * W1b[(64 + k) * 128 + j];
  acc = leaky(acc);
  a1[(size_t)i * 128 + j] = acc;
  atomicAdd(&sum2[j], acc);
  atomicAdd(&sq2[j], acc * acc);
}

__global__ void k_bnstat(const float* __restrict__ sum, const float* __restrict__ sq,
                         const float* __restrict__ gamma, const float* __restrict__ beta,
                         float invCnt, float* __restrict__ sOut, float* __restrict__ oOut) {
  int j = threadIdx.x;
  float m  = sum[j] * invCnt;
  float v  = sq[j] * invCnt - m * m;
  float sc = gamma[j] * rsqrtf(v + 1e-5f);
  sOut[j] = sc;
  oOut[j] = beta[j] - m * sc;
}

__global__ void __launch_bounds__(128) k_g2(
    const float* __restrict__ a1, const float* __restrict__ s, const float* __restrict__ o,
    const float* __restrict__ W, const float* __restrict__ b,
    float* __restrict__ a2, float* __restrict__ sum3, float* __restrict__ sq3) {
  int i = blockIdx.x, j = threadIdx.x;
  __shared__ float hin[128];
  hin[j] = a1[(size_t)i * 128 + j] * s[j] + o[j];
  __syncthreads();
  float acc = b[j];
  for (int k = 0; k < 128; ++k) acc += hin[k] * W[k * 128 + j];
  acc = leaky(acc);
  a2[(size_t)i * 128 + j] = acc;
  atomicAdd(&sum3[j], acc);
  atomicAdd(&sq3[j], acc * acc);
}

__global__ void __launch_bounds__(128) k_g3(
    const float* __restrict__ a2, const float* __restrict__ s, const float* __restrict__ o,
    const float* __restrict__ W, const float* __restrict__ b, float* __restrict__ out) {
  int i = blockIdx.x, j = threadIdx.x;
  __shared__ float hin[128];
  hin[j] = a2[(size_t)i * 128 + j] * s[j] + o[j];
  __syncthreads();
  float acc = b[j];
  for (int k = 0; k < 128; ++k) acc += hin[k] * W[k * 128 + j];
  out[(size_t)i * 128 + j] = acc;
}

// ---------------------------------------------------------------------------
// Launcher
// ---------------------------------------------------------------------------
extern "C" void kernel_launch(void* const* d_in, const int* in_sizes, int n_in,
                              void* d_out, int out_size, void* d_ws, size_t ws_size,
                              hipStream_t stream) {
  const float* x     = (const float*)d_in[0];
  const int*   eidx  = (const int*)d_in[1];    // edge_index [2][E] (int32)
  const float* eattr = (const float*)d_in[2];
  /* d_in[3] = u (unused; gives B) */
  const int*   batch = (const int*)d_in[4];
  const float* W1a = (const float*)d_in[5],  *b1a  = (const float*)d_in[6];
  const float* g1a = (const float*)d_in[7],  *be1a = (const float*)d_in[8];
  const float* W2a = (const float*)d_in[9],  *b2a  = (const float*)d_in[10];
  const float* W1b = (const float*)d_in[11], *b1b  = (const float*)d_in[12];
  const float* g1b = (const float*)d_in[13], *be1b = (const float*)d_in[14];
  const float* W2b = (const float*)d_in[15], *b2b  = (const float*)d_in[16];
  const float* g2b = (const float*)d_in[17], *be2b = (const float*)d_in[18];
  const float* W3b = (const float*)d_in[19], *b3b  = (const float*)d_in[20];

  const int N = in_sizes[0] / 64;
  const int E = in_sizes[2] / 64;
  const int B = in_sizes[3];

  // ---- workspace layout (floats) ----
  float* ws   = (float*)d_ws;
  float* sum1 = ws;               // 128
  float* sq1  = sum1 + 128;       // 128
  float* sum2 = sq1  + 128;       // 128
  float* sq2  = sum2 + 128;       // 128
  float* sum3 = sq2  + 128;       // 128
  float* sq3  = sum3 + 128;       // 128
  float* gxB  = sq3  + 128;       // B*64
  float* gaB  = gxB + (size_t)B * 64;   // B*128
  const int nzero = 768 + B * 192;      // everything above is accumulated into
  float* s2  = gaB + (size_t)B * 128;   // 128
  float* o2  = s2 + 128;
  float* s3  = o2 + 128;
  float* o3  = s3 + 128;
  float* b2P = o3 + 128;                // 128
  float* a1  = b2P + 128;               // B*128
  float* a2  = a1 + (size_t)B * 128;    // B*128
  _Float16* WT1 = (_Float16*)(a2 + (size_t)B * 128);  // 16384 halves (32B aligned)
  _Float16* W2T = WT1 + 16384;                        // 16384 halves

  k_zero<<<(nzero + 255) / 256, 256, 0, stream>>>(ws, nzero);
  k_prepW1<<<64, 256, 0, stream>>>(W1a, WT1);

  const int ntiles = (E + 15) / 16;
  int blocks = (ntiles + 7) / 8;
  if (blocks > 2048) blocks = 2048;
  if (blocks < 1) blocks = 1;

  k_edge1<<<blocks, 256, 0, stream>>>(x, eidx, eattr, b1a, WT1, sum1, sq1, E);
  k_fold1<<<1, 128, 0, stream>>>(sum1, sq1, g1a, be1a, W2a, b2a, 1.0f / (float)E, W2T, b2P);
  k_gx<<<(N * 64 + 255) / 256, 256, 0, stream>>>(x, batch, gxB, N);
  k_edge2<<<blocks, 256, 0, stream>>>(x, eidx, eattr, batch, b1a, WT1, W2T, b2P, gaB, E);

  k_g1<<<B, 128, 0, stream>>>(gxB, gaB, W1b, b1b, a1, sum2, sq2);
  k_bnstat<<<1, 128, 0, stream>>>(sum2, sq2, g1b, be1b, 1.0f / (float)B, s2, o2);
  k_g2<<<B, 128, 0, stream>>>(a1, s2, o2, W2b, b2b, a2, sum3, sq3);
  k_bnstat<<<1, 128, 0, stream>>>(sum3, sq3, g2b, be2b, 1.0f / (float)B, s3, o3);
  k_g3<<<B, 128, 0, stream>>>(a2, s3, o3, W3b, b3b, (float*)d_out);
}
